// S5Block_21380347200326
// MI455X (gfx1250) — matrix operand order repaired and checked by reading, not compiled, not run on hardware
//
#include <hip/hip_runtime.h>
#include <hip/hip_bf16.h>

typedef __bf16 bf16;
typedef __attribute__((ext_vector_type(16))) __bf16 v16bf;
typedef __attribute__((ext_vector_type(8)))  float  v8f;

#define H_DIM 256
#define P_DIM 256
#define BATCH 8
#define SEQ_L 4096
#define M_ROWS (BATCH * SEQ_L)          // 32768
#define M_TILES (M_ROWS / 16)           // 2048
#define N_TILES 16                      // 256/16
#define K_STEPS 8                       // 256/32
#define CHUNK 512
#define NCHUNK (SEQ_L / CHUNK)          // 8
#define STAGE_NT 4                      // N-tiles staged per LDS panel
#define STAGE_ELEMS (STAGE_NT * K_STEPS * 32 * 32)   // 32768 bf16 = 64 KiB

// ---------------------------------------------------------------------------
// Fragment layouts (16x16x32 bf16 WMMA, wave32):
//  A (16x32, M x K): lane l holds row M = l&15; element e holds
//      K = (l>>4)*8 + (e<8 ? e : e+8).
//    A-pack index: ((mt*8 + kk)*32 + lane)*16 + e      (32 B/lane, contiguous)
//  B (32x16, K x N): lane l holds col N = l&15; element e holds
//      K = (l>>4)*8 + (e<8 ? e : e+8).
//    B-pack index (re/im interleaved): ((nt*8 + kk)*32 + l)*32 + e (re), +16 (im)
//  D (16x16 f32): VGPR v holds M = v + 8*(l>>4), N = l&15.
// ---------------------------------------------------------------------------

__device__ inline v8f wmma_bf16(v16bf a, v16bf b, v8f c) {
    return __builtin_amdgcn_wmma_f32_16x16x32_bf16(
        false, a, false, b, (short)0, c, false, false);
}

// ---------------------------------------------------------------------------
// prep: per-p discretization + pack B_bar^T and C^T (re/im interleaved)
// ---------------------------------------------------------------------------
__global__ __launch_bounds__(256) void prep_kernel(
    const float* __restrict__ llr, const float* __restrict__ lim,
    const float* __restrict__ Btr, const float* __restrict__ Bti,
    const float* __restrict__ Ctr, const float* __restrict__ Cti,
    const float* __restrict__ ldl,
    bf16* __restrict__ Bpk, bf16* __restrict__ Cpk,
    float2* __restrict__ lbar)
{
    const int p = threadIdx.x;                  // one thread per state p
    const float lr = -__expf(llr[p]);
    const float li = lim[p];
    const float dt = __expf(ldl[p]);
    const float er = __expf(lr * dt);
    const float abr = er * __cosf(li * dt);
    const float abi = er * __sinf(li * dt);
    lbar[p] = make_float2(abr, abi);

    // factor = (lambda_bar - 1) / lambda
    const float nr = abr - 1.0f, ni = abi;
    const float den = lr * lr + li * li;
    const float fr = (nr * lr + ni * li) / den;
    const float fi = (ni * lr - nr * li) / den;

    // B-pack: GEMM1 B-matrix, (K=h, N=p), value = factor * B_tilde[p, k]
    const int ntB = p >> 4;
    const int lB  = p & 15;
    for (int k = 0; k < H_DIM; ++k) {
        const float br = Btr[p * H_DIM + k], bi = Bti[p * H_DIM + k];
        const float vr = fr * br - fi * bi;
        const float vi = fr * bi + fi * br;
        const int kk = k >> 5, w = k & 31, half = (w >> 3) & 1, e = (w & 7) + ((w & 16) >> 1);
        const int l = lB + half * 16;
        const int idx = ((ntB * 8 + kk) * 32 + l) * 32 + e;
        Bpk[idx]      = (bf16)vr;
        Bpk[idx + 16] = (bf16)vi;
    }

    // C-pack: GEMM2 B-matrix, (K=p fixed = this thread, N=h), value = C[h, p]
    const int kkC = p >> 5, wC = p & 31, halfC = (wC >> 3) & 1, eC = (wC & 7) + ((wC & 16) >> 1);
    for (int hc = 0; hc < H_DIM; ++hc) {
        const int nt = hc >> 4;
        const int l = (hc & 15) + halfC * 16;
        const int idx = ((nt * 8 + kkC) * 32 + l) * 32 + eC;
        Cpk[idx]      = (bf16)Ctr[hc * P_DIM + p];
        Cpk[idx + 16] = (bf16)Cti[hc * P_DIM + p];
    }
}

// ---------------------------------------------------------------------------
// LayerNorm: one wave per row of 256; writes h directly in A-pack order.
// For col c = lane + 32*j: kk = j; within-32 w = lane (constant per lane):
//   half = (lane>>3)&1,  e = (lane&7) + ((lane&16)>>1)
// ---------------------------------------------------------------------------
__global__ __launch_bounds__(256) void ln_kernel(
    const float* __restrict__ u, const float* __restrict__ gamma,
    const float* __restrict__ beta, bf16* __restrict__ hpack)
{
    const int wave = threadIdx.x >> 5;
    const int lane = threadIdx.x & 31;
    const int row = blockIdx.x * 8 + wave;
    const float* ur = u + row * H_DIM;

    float x[8];
    float s = 0.0f, ss = 0.0f;
#pragma unroll
    for (int j = 0; j < 8; ++j) {
        x[j] = ur[lane + 32 * j];
        s += x[j];
        ss += x[j] * x[j];
    }
#pragma unroll
    for (int off = 16; off > 0; off >>= 1) {
        s  += __shfl_xor(s,  off, 32);
        ss += __shfl_xor(ss, off, 32);
    }
    const float mu = s * (1.0f / H_DIM);
    const float var = ss * (1.0f / H_DIM) - mu * mu;
    const float rstd = rsqrtf(var + 1e-5f);

    const int mt = row >> 4;
    const int lg = (row & 15) + (((lane >> 3) & 1) << 4);
    const int e  = (lane & 7) + ((lane & 16) >> 1);
#pragma unroll
    for (int j = 0; j < 8; ++j) {
        const int col = lane + 32 * j;
        const float v = (x[j] - mu) * rstd * gamma[col] + beta[col];
        hpack[((mt * 8 + j) * 32 + lg) * 16 + e] = (bf16)v;
    }
}

// ---------------------------------------------------------------------------
// GEMM1: Bu[m,p] (complex) = h x B_bar^T. One wave per 16-row M-tile; the
// re/im B pack is staged panel-by-panel (4 N-tiles = 64 KiB) through LDS and
// shared by all 8 waves of the block.
// ---------------------------------------------------------------------------
__global__ __launch_bounds__(256) void gemm_bu_kernel(
    const bf16* __restrict__ hpack, const bf16* __restrict__ Bpk,
    float2* __restrict__ Bu)
{
    __shared__ __align__(16) bf16 sB[STAGE_ELEMS];   // 64 KiB panel
    const int tid  = threadIdx.x;
    const int wave = tid >> 5;
    const int lane = tid & 31;
    const int mt = blockIdx.x * 8 + wave;

    // All A fragments for K = 256 held in VGPRs (8 x 32B/lane)
    v16bf A[K_STEPS];
#pragma unroll
    for (int kk = 0; kk < K_STEPS; ++kk)
        A[kk] = *(const v16bf*)(hpack + ((mt * 8 + kk) * 32 + lane) * 16);

    const int ncol  = lane & 15;
    const int mbase = mt * 16 + ((lane >> 4) << 3);

    for (int s = 0; s < N_TILES / STAGE_NT; ++s) {
        const uint4* src = (const uint4*)(Bpk + s * STAGE_ELEMS);
        uint4* dst = (uint4*)sB;
        __syncthreads();                    // previous panel fully consumed
#pragma unroll
        for (int i = 0; i < 16; ++i)
            dst[tid + 256 * i] = src[tid + 256 * i];
        __syncthreads();                    // panel visible to all waves
        if (s < 3)
            __builtin_prefetch(Bpk + (s + 1) * STAGE_ELEMS + tid * 128, 0, 1);

        for (int ntl = 0; ntl < STAGE_NT; ++ntl) {
            v8f accR = {};
            v8f accI = {};
#pragma unroll
            for (int kk = 0; kk < K_STEPS; ++kk) {
                const bf16* fp = sB + ((ntl * 8 + kk) * 32 + lane) * 32;
                v16bf bR = *(const v16bf*)(fp);
                v16bf bI = *(const v16bf*)(fp + 16);
                accR = wmma_bf16(A[kk], bR, accR);
                accI = wmma_bf16(A[kk], bI, accI);
            }
            const int n = (s * STAGE_NT + ntl) * 16 + ncol;
#pragma unroll
            for (int v = 0; v < 8; ++v)
                Bu[(mbase + v) * P_DIM + n] = make_float2(accR[v], accI[v]);
        }
    }
}

// ---------------------------------------------------------------------------
// Chunked scan: block = (batch b, group of 32 p); thread = (p_local, chunk).
// Constant-coefficient recurrence: carry factor a^CHUNK via 9 squarings, one
// barrier resolves all cross-chunk carries. Emits x_re / -x_im in A-pack
// order for GEMM2 (negation folded so GEMM2 is pure accumulate).
// ---------------------------------------------------------------------------
__global__ __launch_bounds__(256) void scan_kernel(
    const float2* __restrict__ Bu, const float2* __restrict__ lbar,
    bf16* __restrict__ xrp, bf16* __restrict__ xip)
{
    __shared__ float2 S[NCHUNK][32];
    const int b  = blockIdx.x >> 3;
    const int pg = blockIdx.x & 7;
    const int chunk = threadIdx.x >> 5;
    const int pl    = threadIdx.x & 31;
    const int p = pg * 32 + pl;

    const float2 a = lbar[p];
    const float ar = a.x, ai = a.y;
    float pr = ar, pi = ai;                    // a^(2^9) = a^CHUNK
#pragma unroll
    for (int i = 0; i < 9; ++i) {
        const float t0 = pr * pr - pi * pi;
        const float t1 = 2.0f * pr * pi;
        pr = t0; pi = t1;
    }

    const int mbase = b * SEQ_L + chunk * CHUNK;
    const int base  = mbase * P_DIM + p;

    // Phase 1: chunk-local scan (from zero) -> chunk sum
    float xrv = 0.0f, xiv = 0.0f;
    for (int t = 0; t < CHUNK; ++t) {
        const float2 bu = Bu[base + t * P_DIM];
        const float nx = ar * xrv - ai * xiv + bu.x;
        const float ny = ar * xiv + ai * xrv + bu.y;
        xrv = nx; xiv = ny;
    }
    S[chunk][pl] = make_float2(xrv, xiv);
    __syncthreads();

    // Phase 2: carry = scan of prior chunk sums with factor a^CHUNK
    float cr = 0.0f, ci = 0.0f;
    for (int j = 0; j < chunk; ++j) {
        const float2 sj = S[j][pl];
        const float t0 = pr * cr - pi * ci + sj.x;
        const float t1 = pr * ci + pi * cr + sj.y;
        cr = t0; ci = t1;
    }

    // Phase 3: re-scan seeded with carry; emit A-pack bf16 operands for GEMM2
    // K dim of GEMM2 is p: kk = p>>5; w = p&31 (constant per thread)
    const int kkp   = p >> 5;
    const int wp    = p & 31;
    const int halfp = (wp >> 3) & 1;
    const int ep    = (wp & 7) + ((wp & 16) >> 1);
    xrv = cr; xiv = ci;
    for (int t = 0; t < CHUNK; ++t) {
        const float2 bu = Bu[base + t * P_DIM];
        const float nx = ar * xrv - ai * xiv + bu.x;
        const float ny = ar * xiv + ai * xrv + bu.y;
        xrv = nx; xiv = ny;
        const int m = mbase + t;
        const int idx = (((m >> 4) * 8 + kkp) * 32 + (m & 15) + halfp * 16) * 16 + ep;
        xrp[idx] = (bf16)xrv;
        xip[idx] = (bf16)(-xiv);
    }
}

__device__ inline float gelu_tanh(float y) {
    const float t = 0.7978845608028654f * (y + 0.044715f * y * y * y);
    return 0.5f * y * (1.0f + tanhf(t));
}

// ---------------------------------------------------------------------------
// GEMM2 + epilogue: y = x_re*C_re^T + (-x_im)*C_im^T ; out = u + gelu(y)
// Same LDS panel staging for the C pack.
// ---------------------------------------------------------------------------
__global__ __launch_bounds__(256) void gemm_y_kernel(
    const bf16* __restrict__ xrp, const bf16* __restrict__ xip,
    const bf16* __restrict__ Cpk,
    const float* __restrict__ u, float* __restrict__ out)
{
    __shared__ __align__(16) bf16 sB[STAGE_ELEMS];   // 64 KiB panel
    const int tid  = threadIdx.x;
    const int wave = tid >> 5;
    const int lane = tid & 31;
    const int mt = blockIdx.x * 8 + wave;

    v16bf Ar[K_STEPS], Ai[K_STEPS];
#pragma unroll
    for (int kk = 0; kk < K_STEPS; ++kk) {
        Ar[kk] = *(const v16bf*)(xrp + ((mt * 8 + kk) * 32 + lane) * 16);
        Ai[kk] = *(const v16bf*)(xip + ((mt * 8 + kk) * 32 + lane) * 16);
    }

    const int ncol  = lane & 15;
    const int mbase = mt * 16 + ((lane >> 4) << 3);

    for (int s = 0; s < N_TILES / STAGE_NT; ++s) {
        const uint4* src = (const uint4*)(Cpk + s * STAGE_ELEMS);
        uint4* dst = (uint4*)sB;
        __syncthreads();
#pragma unroll
        for (int i = 0; i < 16; ++i)
            dst[tid + 256 * i] = src[tid + 256 * i];
        __syncthreads();
        if (s < 3)
            __builtin_prefetch(Cpk + (s + 1) * STAGE_ELEMS + tid * 128, 0, 1);

        for (int ntl = 0; ntl < STAGE_NT; ++ntl) {
            v8f acc = {};
#pragma unroll
            for (int kk = 0; kk < K_STEPS; ++kk) {
                const bf16* fp = sB + ((ntl * 8 + kk) * 32 + lane) * 32;
                v16bf cR = *(const v16bf*)(fp);
                v16bf cI = *(const v16bf*)(fp + 16);
                acc = wmma_bf16(Ar[kk], cR, acc);
                acc = wmma_bf16(Ai[kk], cI, acc);
            }
            const int n = (s * STAGE_NT + ntl) * 16 + ncol;
#pragma unroll
            for (int v = 0; v < 8; ++v) {
                const int idx = (mbase + v) * H_DIM + n;
                out[idx] = u[idx] + gelu_tanh(acc[v]);
            }
        }
    }
}

// ---------------------------------------------------------------------------
extern "C" void kernel_launch(void* const* d_in, const int* in_sizes, int n_in,
                              void* d_out, int out_size, void* d_ws, size_t ws_size,
                              hipStream_t stream)
{
    const float* u   = (const float*)d_in[0];
    const float* llr = (const float*)d_in[1];
    const float* lim = (const float*)d_in[2];
    const float* Btr = (const float*)d_in[3];
    const float* Bti = (const float*)d_in[4];
    const float* Ctr = (const float*)d_in[5];
    const float* Cti = (const float*)d_in[6];
    const float* ldl = (const float*)d_in[7];
    const float* gam = (const float*)d_in[8];
    const float* bet = (const float*)d_in[9];
    float* out = (float*)d_out;

    char* ws = (char*)d_ws;
    const size_t MB = 1024 * 1024;
    bf16*   hpack = (bf16*)(ws);                        // 16 MiB (A-pack order)
    bf16*   xrp   = (bf16*)(ws + 16 * MB);              // 16 MiB
    bf16*   xip   = (bf16*)(ws + 32 * MB);              // 16 MiB
    float2* Bu    = (float2*)(ws + 48 * MB);            // 64 MiB
    bf16*   Bpk   = (bf16*)(ws + 112 * MB);             // 256 KiB (re/im interleaved)
    bf16*   Cpk   = (bf16*)(ws + 112 * MB + 256 * 1024);// 256 KiB
    float2* lbar  = (float2*)(ws + 112 * MB + 512 * 1024);

    prep_kernel<<<1, 256, 0, stream>>>(llr, lim, Btr, Bti, Ctr, Cti, ldl,
                                       Bpk, Cpk, lbar);
    ln_kernel<<<M_ROWS / 8, 256, 0, stream>>>(u, gam, bet, hpack);
    gemm_bu_kernel<<<M_TILES / 8, 256, 0, stream>>>(hpack, Bpk, Bu);
    scan_kernel<<<BATCH * (P_DIM / 32), 256, 0, stream>>>(Bu, lbar, xrp, xip);
    gemm_y_kernel<<<M_TILES / 8, 256, 0, stream>>>(xrp, xip, Cpk, u, out);
}
